// GCN_76836964925991
// MI455X (gfx1250) — matrix operand order repaired and checked by reading, not compiled, not run on hardware
//
#include <hip/hip_runtime.h>
#include <hip/hip_bf16.h>

// ---------------------------------------------------------------------------
// GCN model for MI455X (gfx1250, wave32, WMMA).
// Hot path: fused FFN (x@W1 -> BN/ReLU -> @W2 -> BN), 98.3 GFLOP, done as
// f16-in/f32-acc WMMA with the [64000,3840] intermediate kept in LDS/regs.
// Each wave owns M=32 rows so every B fragment feeds 2 WMMAs.
// GCN layers: VALU GEMM (weights staged to LDS via async-load-to-LDS) +
// global_atomic_add_f32 scatter.
// ---------------------------------------------------------------------------

typedef __attribute__((ext_vector_type(16))) _Float16 v16h;
typedef __attribute__((ext_vector_type(8)))  float    v8f;

#define NNODES   64000          // 64 graphs * 1000 nodes
#define NEDGES   2048000        // NNODES * 32
#define FIN      100
#define K1PAD    128            // FIN padded to 128 (4 K-chunks of 32)
#define N1       3840
#define NT1      240            // N1/16
#define KC2      120            // 3840/32 (K-chunks of GEMM2)
#define NT2      7              // ceil(100/16)
#define EPS      1e-5f

// ---- optional gfx1250 async LDS-direct loads (guarded; safe fallback) -----
#if defined(__has_builtin)
#if __has_builtin(__builtin_amdgcn_global_load_async_to_lds_b32)
#define HAVE_ASYNC_LDS 1
#endif
#endif

#ifdef HAVE_ASYNC_LDS
typedef __attribute__((address_space(1))) int g_int;
typedef __attribute__((address_space(3))) int l_int;
#if __has_builtin(__builtin_amdgcn_s_wait_asynccnt)
#define ASYNC_WAIT() __builtin_amdgcn_s_wait_asynccnt(0)
#else
#define ASYNC_WAIT() asm volatile("s_wait_asynccnt 0" ::: "memory")
#endif
#endif

// ---------------------------- packing kernels ------------------------------

// x[64000,100] f32 -> Ax[64000,128] f16 (K padded with zeros, row major).
__global__ __launch_bounds__(256) void pack_x(const float* __restrict__ x,
                                              _Float16* __restrict__ Ax) {
    size_t i = (size_t)blockIdx.x * 256 + threadIdx.x;
    if (i >= (size_t)NNODES * K1PAD) return;
    int r = (int)(i >> 7), k = (int)(i & 127);
    Ax[i] = (_Float16)(k < FIN ? x[(size_t)r * FIN + k] : 0.f);
}

// W1[100,3840] -> per-lane WMMA B fragments.
// elem index = ((nt*4 + kci)*32 + lane)*16 + j,
//   n = nt*16 + (lane&15),  k = kci*32 + (lane>>4)*16 + j.
__global__ __launch_bounds__(256) void pack_w1(const float* __restrict__ W1,
                                               _Float16* __restrict__ W1p) {
    size_t i = (size_t)blockIdx.x * 256 + threadIdx.x;
    if (i >= (size_t)NT1 * 4 * 32 * 16) return;
    int j = (int)(i & 15);
    size_t s = i >> 4;
    int lane = (int)(s & 31);
    size_t s2 = s >> 5;
    int kci = (int)(s2 & 3);
    int nt  = (int)(s2 >> 2);
    int n = nt * 16 + (lane & 15);
    int k = kci * 32 + (lane >> 4) * 16 + j;
    W1p[i] = (_Float16)(k < FIN ? W1[(size_t)k * N1 + n] : 0.f);
}

// W2[3840,100] -> fragments: elem = ((nt2*KC2 + kc2)*32 + lane)*16 + j.
__global__ __launch_bounds__(256) void pack_w2(const float* __restrict__ W2,
                                               _Float16* __restrict__ W2p) {
    size_t i = (size_t)blockIdx.x * 256 + threadIdx.x;
    if (i >= (size_t)NT2 * KC2 * 32 * 16) return;
    int j = (int)(i & 15);
    size_t s = i >> 4;
    int lane = (int)(s & 31);
    size_t s2 = s >> 5;
    int kc2 = (int)(s2 % KC2);
    int nt2 = (int)(s2 / KC2);
    int n = nt2 * 16 + (lane & 15);
    int k = kc2 * 32 + (lane >> 4) * 16 + j;
    W2p[i] = (_Float16)(n < FIN ? W2[(size_t)k * FIN + n] : 0.f);
}

// ------------------------- fused FFN (the hot path) ------------------------
// One wave per 32-row block (two 16-row WMMA tiles). Each GEMM1 output tile
// streams through a 2KB LDS staging buffer that re-shapes WMMA C-layout into
// A-layout for GEMM2; every B fragment is reused for both row tiles.
__global__ __launch_bounds__(32)
void ffn_fused(const _Float16* __restrict__ Ax,
               const _Float16* __restrict__ W1p,
               const _Float16* __restrict__ W2p,
               const float* __restrict__ b1, const float* __restrict__ g1,
               const float* __restrict__ be1, const float* __restrict__ m1,
               const float* __restrict__ v1,
               const float* __restrict__ b2, const float* __restrict__ g2,
               const float* __restrict__ be2, const float* __restrict__ m2,
               const float* __restrict__ v2,
               float* __restrict__ h2) {
    __shared__ __align__(16) _Float16 tile[2][16 * 32];
    const int lane = threadIdx.x;
    const int m0   = blockIdx.x * 32;
    const int row  = lane & 15;
    const int kh   = lane >> 4;

    // A1 fragments for both row tiles (4 K-chunks each), resident in VGPRs.
    v16h a1[2][4];
#pragma unroll
    for (int b = 0; b < 2; ++b) {
        const _Float16* arow = Ax + (size_t)(m0 + b * 16 + row) * K1PAD;
#pragma unroll
        for (int c = 0; c < 4; ++c) {
            ((uint4*)&a1[b][c])[0] = *(const uint4*)(arow + c * 32 + kh * 16);
            ((uint4*)&a1[b][c])[1] = *(const uint4*)(arow + c * 32 + kh * 16 + 8);
        }
    }

    v8f acc2[2][NT2];
#pragma unroll
    for (int b = 0; b < 2; ++b)
#pragma unroll
        for (int t = 0; t < NT2; ++t) acc2[b][t] = (v8f){};

    for (int kc2 = 0; kc2 < KC2; ++kc2) {
        // --- produce one 16x32 A2 chunk per row tile = two GEMM1 tiles ---
#pragma unroll
        for (int half = 0; half < 2; ++half) {
            const int nt1 = kc2 * 2 + half;
            v8f c1[2];
            c1[0] = (v8f){};
            c1[1] = (v8f){};
#pragma unroll
            for (int kci = 0; kci < 4; ++kci) {
                const _Float16* bp =
                    W1p + ((size_t)(nt1 * 4 + kci) * 32 + lane) * 16;
                v16h bfrag;
                ((uint4*)&bfrag)[0] = *(const uint4*)(bp);
                ((uint4*)&bfrag)[1] = *(const uint4*)(bp + 8);
#pragma unroll
                for (int b = 0; b < 2; ++b)
                    c1[b] = __builtin_amdgcn_wmma_f32_16x16x32_f16(
                        false, a1[b][kci], false, bfrag, (short)0, c1[b],
                        false, false);
            }
            // BN1 + ReLU epilogue, write f16 into staging tiles.
            const int ncol = lane & 15;
            const int n    = nt1 * 16 + ncol;        // GEMM1 col == GEMM2 k
            const float sc = g1[n] * __frsqrt_rn(v1[n] + EPS);
            const float sh = be1[n] - m1[n] * sc;
            const float bb = b1[n];
            const int mbase = kh * 8;
#pragma unroll
            for (int b = 0; b < 2; ++b)
#pragma unroll
                for (int r = 0; r < 8; ++r) {
                    float y = (c1[b][r] + bb) * sc + sh;
                    y = y > 0.f ? y : 0.f;
                    tile[b][(mbase + r) * 32 + half * 16 + ncol] = (_Float16)y;
                }
        }
        __syncthreads();

        // Read back as A-layout fragments.
        v16h a2[2];
#pragma unroll
        for (int b = 0; b < 2; ++b) {
            const _Float16* t0 = tile[b] + row * 32 + kh * 16;
            ((uint4*)&a2[b])[0] = *(const uint4*)(t0);
            ((uint4*)&a2[b])[1] = *(const uint4*)(t0 + 8);
        }

        if (kc2 + 1 < KC2)   // warm caches for the next W2 fragment column
            __builtin_prefetch(W2p + ((size_t)(kc2 + 1) * 32 + lane) * 16, 0, 3);

#pragma unroll
        for (int t = 0; t < NT2; ++t) {
            const _Float16* bp =
                W2p + ((size_t)(t * KC2 + kc2) * 32 + lane) * 16;
            v16h bfrag;
            ((uint4*)&bfrag)[0] = *(const uint4*)(bp);
            ((uint4*)&bfrag)[1] = *(const uint4*)(bp + 8);
#pragma unroll
            for (int b = 0; b < 2; ++b)
                acc2[b][t] = __builtin_amdgcn_wmma_f32_16x16x32_f16(
                    false, a2[b], false, bfrag, (short)0, acc2[b][t],
                    false, false);
        }
        __syncthreads();
    }

    // BN2 epilogue (no relu), store f32 h2[64000,100].
    const int ncol  = lane & 15;
    const int mbase = kh * 8;
#pragma unroll
    for (int t = 0; t < NT2; ++t) {
        const int n = t * 16 + ncol;
        if (n < FIN) {
            const float sc = g2[n] * __frsqrt_rn(v2[n] + EPS);
            const float sh = be2[n] - m2[n] * sc;
            const float bb = b2[n];
#pragma unroll
            for (int b = 0; b < 2; ++b)
#pragma unroll
                for (int r = 0; r < 8; ++r) {
                    float y = (acc2[b][t][r] + bb) * sc + sh;
                    h2[(size_t)(m0 + b * 16 + mbase + r) * FIN + n] = y;
                }
        }
    }
}

// ----------------------------- GCN machinery -------------------------------

__global__ __launch_bounds__(256) void deg_init(float* deg) {
    int n = blockIdx.x * 256 + threadIdx.x;
    if (n < NNODES) deg[n] = 1.f;                    // self loop
}
__global__ __launch_bounds__(256) void deg_count(const int* __restrict__ dst,
                                                 float* deg) {
    int e = blockIdx.x * 256 + threadIdx.x;
    if (e < NEDGES) atomicAdd(&deg[dst[e]], 1.f);
}
__global__ __launch_bounds__(256) void deg_inv(float* deg) {
    int n = blockIdx.x * 256 + threadIdx.x;
    if (n < NNODES) deg[n] = __frsqrt_rn(deg[n]);
}

// H[n,:] = X[n,:] @ W  (weights staged into LDS; async LDS-direct when avail)
__global__ __launch_bounds__(256)
void node_gemm(const float* __restrict__ X, const float* __restrict__ W,
               float* __restrict__ H, int Fin, int Fout) {
    __shared__ float Ws[FIN * 64];
#ifdef HAVE_ASYNC_LDS
    for (int i = threadIdx.x; i < Fin * Fout; i += 256)
        __builtin_amdgcn_global_load_async_to_lds_b32(
            (g_int*)(W + i), (l_int*)(&Ws[i]), 0, 0);
    ASYNC_WAIT();
    __syncthreads();
#else
    for (int i = threadIdx.x; i < Fin * Fout; i += 256) Ws[i] = W[i];
    __syncthreads();
#endif
    int n = blockIdx.x * 256 + threadIdx.x;
    if (n >= NNODES) return;
    const float* xr = X + (size_t)n * Fin;
    float* hr = H + (size_t)n * Fout;
    for (int fo = 0; fo < Fout; ++fo) {
        float acc = 0.f;
        for (int fi = 0; fi < Fin; ++fi) acc += xr[fi] * Ws[fi * Fout + fo];
        hr[fo] = acc;
    }
}

// AGG = H * dinv^2 (self-loop term) + bias
__global__ __launch_bounds__(256)
void agg_init(const float* __restrict__ H, const float* __restrict__ dinv,
              const float* __restrict__ bias, float* __restrict__ AGG,
              int Fout) {
    size_t i = (size_t)blockIdx.x * 256 + threadIdx.x;
    if (i >= (size_t)NNODES * Fout) return;
    int n = (int)(i / Fout), f = (int)(i % Fout);
    float d = dinv[n];
    AGG[i] = H[i] * d * d + bias[f];
}

// AGG[dst] += H[src] * dinv[src]*dinv[dst]   (segment_sum via atomics)
__global__ __launch_bounds__(256)
void gcn_scatter(const int* __restrict__ src, const int* __restrict__ dst,
                 const float* __restrict__ H, const float* __restrict__ dinv,
                 float* __restrict__ AGG, int Fout) {
    int e = blockIdx.x * 256 + threadIdx.x;
    if (e >= NEDGES) return;
    int s = src[e], d = dst[e];
    float c = dinv[s] * dinv[d];
    const float* hs = H + (size_t)s * Fout;
    float* ad = AGG + (size_t)d * Fout;
    for (int f = 0; f < Fout; ++f) atomicAdd(&ad[f], hs[f] * c);
}

__global__ __launch_bounds__(256) void relu_ip(float* A, size_t total) {
    size_t i = (size_t)blockIdx.x * 256 + threadIdx.x;
    if (i < total) A[i] = fmaxf(A[i], 0.f);
}

// out[g,c] = bf[c] + sum_j h5flat[g*4000+j] * Wf[j,c]
__global__ __launch_bounds__(256)
void final_fc(const float* __restrict__ Hf, const float* __restrict__ Wf,
              const float* __restrict__ bf, float* __restrict__ out) {
    __shared__ float red[512];
    int g = blockIdx.x, tid = threadIdx.x;
    float a0 = 0.f, a1 = 0.f;
    for (int j = tid; j < 4000; j += 256) {
        float hv = Hf[(size_t)g * 4000 + j];   // [64000,4] flat == [64,4000]
        a0 += hv * Wf[j * 2 + 0];
        a1 += hv * Wf[j * 2 + 1];
    }
    red[tid] = a0; red[256 + tid] = a1;
    __syncthreads();
    for (int s = 128; s > 0; s >>= 1) {
        if (tid < s) { red[tid] += red[tid + s]; red[256 + tid] += red[256 + tid + s]; }
        __syncthreads();
    }
    if (tid == 0) {
        out[g * 2 + 0] = red[0]   + bf[0];
        out[g * 2 + 1] = red[256] + bf[1];
    }
}

// ------------------------------- launcher ----------------------------------

extern "C" void kernel_launch(void* const* d_in, const int* in_sizes, int n_in,
                              void* d_out, int out_size, void* d_ws, size_t ws_size,
                              hipStream_t stream) {
    const float* x  = (const float*)d_in[0];
    const int*   ei = (const int*)d_in[1];
    const int*   src = ei;
    const int*   dst = ei + NEDGES;
    const float* W1 = (const float*)d_in[2];
    const float* b1 = (const float*)d_in[3];
    const float* g1 = (const float*)d_in[4];
    const float* be1 = (const float*)d_in[5];
    const float* m1 = (const float*)d_in[6];
    const float* v1 = (const float*)d_in[7];
    const float* W2 = (const float*)d_in[8];
    const float* b2 = (const float*)d_in[9];
    const float* g2 = (const float*)d_in[10];
    const float* be2 = (const float*)d_in[11];
    const float* m2 = (const float*)d_in[12];
    const float* v2 = (const float*)d_in[13];
    const float* Wf = (const float*)d_in[14];
    const float* bf = (const float*)d_in[15];
    const float* Wc[5] = {(const float*)d_in[16], (const float*)d_in[18],
                          (const float*)d_in[20], (const float*)d_in[22],
                          (const float*)d_in[24]};
    const float* bc[5] = {(const float*)d_in[17], (const float*)d_in[19],
                          (const float*)d_in[21], (const float*)d_in[23],
                          (const float*)d_in[25]};
    float* out = (float*)d_out;

    // Workspace carve-up (all regions fully written before read).
    char* ws = (char*)d_ws;
    size_t off = 0;
    auto carve = [&](size_t bytes) -> char* {
        char* p = ws + off;
        off = (off + bytes + 255) & ~(size_t)255;
        return p;
    };
    _Float16* Ax  = (_Float16*)carve((size_t)NNODES * K1PAD * 2);
    _Float16* W1p = (_Float16*)carve((size_t)NT1 * 4 * 512 * 2);
    _Float16* W2p = (_Float16*)carve((size_t)NT2 * KC2 * 512 * 2);
    float* h2   = (float*)carve((size_t)NNODES * FIN * 4);
    float* dinv = (float*)carve((size_t)NNODES * 4);
    float* Hbuf = (float*)carve((size_t)NNODES * 64 * 4);
    float* AGGa = (float*)carve((size_t)NNODES * 64 * 4);
    float* AGGb = (float*)carve((size_t)NNODES * 64 * 4);
    (void)ws_size; (void)n_in; (void)in_sizes; (void)out_size;

    // 1) pack operands into f16 / WMMA fragment layouts
    pack_x <<<(NNODES * K1PAD + 255) / 256, 256, 0, stream>>>(x, Ax);
    pack_w1<<<(NT1 * 4 * 512 + 255) / 256, 256, 0, stream>>>(W1, W1p);
    pack_w2<<<(NT2 * KC2 * 512 + 255) / 256, 256, 0, stream>>>(W2, W2p);

    // 2) fused FFN: one wave per 32-row block
    ffn_fused<<<NNODES / 32, 32, 0, stream>>>(Ax, W1p, W2p,
                                              b1, g1, be1, m1, v1,
                                              b2, g2, be2, m2, v2, h2);

    // 3) degrees (shared across all 5 GCN layers)
    deg_init <<<NNODES / 256, 256, 0, stream>>>(dinv);
    deg_count<<<(NEDGES + 255) / 256, 256, 0, stream>>>(dst, dinv);
    deg_inv  <<<NNODES / 256, 256, 0, stream>>>(dinv);

    // 4) five GCN layers
    const int Fi[5] = {100, 64, 32, 16, 8};
    const int Fo[5] = {64, 32, 16, 8, 4};
    float* X = h2;
    float* aggbuf[2] = {AGGa, AGGb};
    for (int L = 0; L < 5; ++L) {
        float* A = aggbuf[L & 1];
        size_t tot = (size_t)NNODES * Fo[L];
        node_gemm  <<<NNODES / 256, 256, 0, stream>>>(X, Wc[L], Hbuf, Fi[L], Fo[L]);
        agg_init   <<<(tot + 255) / 256, 256, 0, stream>>>(Hbuf, dinv, bc[L], A, Fo[L]);
        gcn_scatter<<<(NEDGES + 255) / 256, 256, 0, stream>>>(src, dst, Hbuf, dinv, A, Fo[L]);
        relu_ip    <<<(tot + 255) / 256, 256, 0, stream>>>(A, tot);
        X = A;
    }

    // 5) readout
    final_fc<<<64, 256, 0, stream>>>(X, Wf, bf, out);
}
